// EagerBidirectionalAttentionBlock_44100724195899
// MI455X (gfx1250) — matrix operand order repaired and checked by reading, not compiled, not run on hardware
//
#include <hip/hip_runtime.h>
#include <math.h>

// ---------------------------------------------------------------------------
// EagerBidirectionalAttentionBlock for MI455X (gfx1250, wave32, WMMA bf16)
//   B=4, S=2048, H=768, NH=12, HD=64
// Pipeline:
//   1) f32 -> bf16 convert (x, wq, wk, wv, wo)
//   2) QKV projection GEMMs: TDM (tensor_load_to_lds) double-buffered panels
//      + v_wmma_f32_16x16x32_bf16; V stored transposed for the PV GEMM
//   3) Flash attention (online softmax), K/V tiles TDM-staged into LDS
//   4) Output projection GEMM (+bias) -> f32
// ---------------------------------------------------------------------------

#define DEVINL static __device__ __forceinline__

typedef unsigned short u16;
typedef __bf16 bf16;
typedef bf16  bf16x16 __attribute__((ext_vector_type(16)));
typedef float f32x8   __attribute__((ext_vector_type(8)));
typedef unsigned int u32x4 __attribute__((ext_vector_type(4)));
typedef int i32x4 __attribute__((ext_vector_type(4)));
typedef int i32x8 __attribute__((ext_vector_type(8)));

constexpr int Bc = 4, Sc = 2048, Hc = 768, NHc = 12, HDc = 64;
constexpr int Mc = Bc * Sc;                 // 8192 rows for the big GEMMs
constexpr float NEG_INF = -3.0e38f;
constexpr float SCALE = 0.125f;             // 1/sqrt(64)

DEVINL u16 f2bf(float f) {                  // round-to-nearest-even f32->bf16
  unsigned int u = __float_as_uint(f);
  u += 0x7FFFu + ((u >> 16) & 1u);
  return (u16)(u >> 16);
}

DEVINL f32x8 zero8() {
  f32x8 z;
#pragma unroll
  for (int i = 0; i < 8; ++i) z[i] = 0.0f;
  return z;
}

union TileU { bf16x16 v; int4 q[2]; };

// A-matrix 16x32 bf16 tile (CDNA5 ISA 7.12.2):
//   lane L: M = L&15; VGPRs 0-3 hold K = (L>=16?8:0)+0..7, VGPRs 4-7 hold +16.
DEVINL bf16x16 load_a_tile(const u16* p, int ld, int row0, int col0, int r, int half) {
  const u16* base = p + (size_t)(row0 + r) * ld + col0 + half * 8;
  TileU t;
  t.q[0] = *(const int4*)(base);
  t.q[1] = *(const int4*)(base + 16);
  return t.v;
}

// B-matrix 32x16 bf16 tile (column-major in memory == provider rows):
//   lane L: N = L&15; VGPRs 0-3 hold K = (L>=16?8:0)+0..7, VGPRs 4-7 hold +16.
DEVINL bf16x16 load_b_tile(const u16* p, int ld, int col_row0, int k0, int r, int half) {
  const u16* base = p + (size_t)(col_row0 + r) * ld + k0 + half * 8;
  TileU t;
  t.q[0] = *(const int4*)(base);
  t.q[1] = *(const int4*)(base + 16);
  return t.v;
}

DEVINL f32x8 wmma_bf16(bf16x16 a, bf16x16 b, f32x8 c) {
  return __builtin_amdgcn_wmma_f32_16x16x32_bf16(false, a, false, b, (short)0, c,
                                                 false, false);
}

// ---------------------------------------------------------------------------
// Tensor Data Mover: DMA a [tile_rows x tile_cols] bf16 tile (row stride
// tensor_ld elements) from global memory into LDS at byte offset lds_off.
// D# packing per CDNA5 ISA 8.3/8.4 (2-D tensor, groups 2/3 zero).
// Wave-level op (EXEC ignored); tracked by the issuing wave's TENSORcnt.
// This toolchain exposes the 6-arg builtin form (extra zero i32x8 group).
// ---------------------------------------------------------------------------
DEVINL void tdm_load_2d_bf16(unsigned lds_off, const u16* gsrc, int tensor_ld,
                             int tensor_rows, int tile_cols, int tile_rows) {
  unsigned long long ga = (unsigned long long)(uintptr_t)gsrc;

  u32x4 g0;
  g0[0] = 1u;                                  // count=1 (valid), no gather
  g0[1] = lds_off;                             // LDS byte address
  g0[2] = (unsigned)ga;                        // global_addr[31:0]
  g0[3] = (unsigned)((ga >> 32) & 0x01FFFFFFu) | (2u << 30);  // addr[56:32], type=2

  const unsigned long long td0 = (unsigned)tensor_ld;    // tensor_dim0 (elems)
  const unsigned long long td1 = (unsigned)tensor_rows;  // tensor_dim1
  const unsigned long long tl0 = (unsigned)tile_cols;    // tile_dim0
  const unsigned long long tl1 = (unsigned)tile_rows;    // tile_dim1
  const unsigned long long st0 = (unsigned)tensor_ld;    // tensor_dim0_stride

  // group1 bit layout: [17:16]=data_size(1 => 2B)  [79:48]=tensor_dim0
  // [111:80]=tensor_dim1 [127:112]=tile_dim0 [143:128]=tile_dim1
  // [159:144]=tile_dim2(0) [207:160]=tensor_dim0_stride [255:208]=dim1_stride(0)
  const unsigned long long qw0 = (1ull << 16) | (td0 << 48);
  const unsigned long long qw1 = (td0 >> 16) | (td1 << 16) | (tl0 << 48);
  const unsigned long long qw2 = tl1 | ((st0 & 0xFFFFFFFFull) << 32);
  const unsigned long long qw3 = (st0 >> 32);

  i32x8 g1;
  g1[0] = (int)(unsigned)qw0; g1[1] = (int)(unsigned)(qw0 >> 32);
  g1[2] = (int)(unsigned)qw1; g1[3] = (int)(unsigned)(qw1 >> 32);
  g1[4] = (int)(unsigned)qw2; g1[5] = (int)(unsigned)(qw2 >> 32);
  g1[6] = (int)(unsigned)qw3; g1[7] = (int)(unsigned)(qw3 >> 32);

  i32x4 gz4 = {0, 0, 0, 0};
  i32x8 gz8 = {0, 0, 0, 0, 0, 0, 0, 0};
  __builtin_amdgcn_tensor_load_to_lds(g0, g1, gz4, gz4, gz8, 0);
}

DEVINL void tdm_wait() { __builtin_amdgcn_s_wait_tensorcnt(0); }

DEVINL unsigned lds_addr_of(const void* p) {   // LDS byte offset from flat addr
  return (unsigned)(uintptr_t)p;
}

// ---------------------------------------------------------------------------
// Kernel 1: f32 -> bf16 conversion (vectorized x4)
// ---------------------------------------------------------------------------
__global__ __launch_bounds__(256) void f32_to_bf16_kernel(const float* __restrict__ src,
                                                          u16* __restrict__ dst, int n4) {
  int i = blockIdx.x * blockDim.x + threadIdx.x;
  if (i < n4) {
    float4 f = ((const float4*)src)[i];
    ushort4 o;
    o.x = f2bf(f.x); o.y = f2bf(f.y); o.z = f2bf(f.z); o.w = f2bf(f.w);
    ((ushort4*)dst)[i] = o;
  }
}

// ---------------------------------------------------------------------------
// Kernel 2: fused QKV projection.  out = x @ W^T + b   (M=8192, N=768, K=768)
// blockIdx.z: 0->Q, 1->K (bf16 [M][768]), 2->V (bf16 transposed [B][NH][HD][S])
// Block: 256 thr = 8 waves; wave tile 32x64; block tile 128x128.
// A/B panels (128x32 bf16) TDM-staged into LDS, double-buffered.
// ---------------------------------------------------------------------------
__global__ __launch_bounds__(256) void qkv_gemm_kernel(
    const u16* __restrict__ xbf, const u16* __restrict__ wbf,
    const float* __restrict__ bq, const float* __restrict__ bk,
    const float* __restrict__ bv,
    u16* __restrict__ qbf, u16* __restrict__ kbf, u16* __restrict__ vtb) {
  __shared__ u16 Ash[2][128 * 32];   // 8 KB each buffer
  __shared__ u16 Bsh[2][128 * 32];

  const int z = blockIdx.z;
  const u16* W = wbf + (size_t)z * Hc * Hc;
  const float* bias = (z == 0) ? bq : (z == 1) ? bk : bv;

  const int wave = threadIdx.x >> 5;
  const int lane = threadIdx.x & 31;
  const int r = lane & 15, half = lane >> 4;
  const int blkM = blockIdx.x * 128;
  const int blkN = blockIdx.y * 128;
  const int mb = (wave >> 1) * 32;   // wave tile origin inside the LDS panel
  const int nb = (wave & 1) * 64;

  // prologue: stage 0
  if (wave == 0)
    tdm_load_2d_bf16(lds_addr_of(&Ash[0][0]), xbf + (size_t)blkM * Hc, Hc, Mc, 32, 128);
  else if (wave == 1)
    tdm_load_2d_bf16(lds_addr_of(&Bsh[0][0]), W + (size_t)blkN * Hc, Hc, Hc, 32, 128);
  if (wave < 2) tdm_wait();
  __syncthreads();

  f32x8 acc[2][4];
#pragma unroll
  for (int i = 0; i < 2; ++i)
#pragma unroll
    for (int j = 0; j < 4; ++j) acc[i][j] = zero8();

  for (int kk = 0, stage = 0; kk < Hc; kk += 32, ++stage) {
    const int buf = stage & 1;
    // kick off DMA for next k-panel while this one computes
    if (kk + 32 < Hc) {
      if (wave == 0)
        tdm_load_2d_bf16(lds_addr_of(&Ash[buf ^ 1][0]),
                         xbf + (size_t)blkM * Hc + kk + 32, Hc, Mc, 32, 128);
      else if (wave == 1)
        tdm_load_2d_bf16(lds_addr_of(&Bsh[buf ^ 1][0]),
                         W + (size_t)blkN * Hc + kk + 32, Hc, Hc, 32, 128);
    }

    bf16x16 a0 = load_a_tile(&Ash[buf][0], 32, mb, 0, r, half);
    bf16x16 a1 = load_a_tile(&Ash[buf][0], 32, mb + 16, 0, r, half);
#pragma unroll
    for (int j = 0; j < 4; ++j) {
      bf16x16 bt = load_b_tile(&Bsh[buf][0], 32, nb + j * 16, 0, r, half);
      acc[0][j] = wmma_bf16(a0, bt, acc[0][j]);
      acc[1][j] = wmma_bf16(a1, bt, acc[1][j]);
    }

    if (wave < 2) tdm_wait();   // next panel landed
    __syncthreads();            // all readers done + next panel visible
  }

#pragma unroll
  for (int i = 0; i < 2; ++i) {
#pragma unroll
    for (int j = 0; j < 4; ++j) {
      const int col = blkN + nb + j * 16 + r;
      const float bcol = bias[col];
#pragma unroll
      for (int rr = 0; rr < 8; ++rr) {
        const int row = blkM + mb + i * 16 + rr + half * 8;
        const u16 v = f2bf(acc[i][j][rr] + bcol);
        if (z == 0) {
          qbf[(size_t)row * Hc + col] = v;
        } else if (z == 1) {
          kbf[(size_t)row * Hc + col] = v;
        } else {
          const int bb = row >> 11, s = row & (Sc - 1);
          const int hh = col >> 6,  d = col & (HDc - 1);
          vtb[(((size_t)(bb * NHc + hh) * HDc + d) * Sc) + s] = v;
        }
      }
    }
  }
}

// ---------------------------------------------------------------------------
// Kernel 3: flash attention.  Each wave: one 16-query tile; block shares a
// TDM-staged 32-key K tile (32x64) and V^T tile (64x32), double-buffered.
// Grid (S/16/8, NH, B), 256 thr = 8 waves.  LDS also relayouts P (C->A).
// ---------------------------------------------------------------------------
__global__ __launch_bounds__(256) void attn_kernel(
    const u16* __restrict__ qbf, const u16* __restrict__ kbf,
    const u16* __restrict__ vtb, const unsigned char* __restrict__ mask8,
    u16* __restrict__ obf) {
  __shared__ u16 psh[8][16 * 32];   // per-wave P tile (1 KB each)
  __shared__ u16 Ksh[2][32 * 64];   // [key][d]  4 KB each buffer
  __shared__ u16 Vsh[2][64 * 32];   // [d][key]  4 KB each buffer

  const int b = blockIdx.z, h = blockIdx.y;
  const int wave = threadIdx.x >> 5, lane = threadIdx.x & 31;
  const int r = lane & 15, half = lane >> 4;
  const int sb = (blockIdx.x * 8 + wave) * 16;   // query-row base in [0,S)
  const int qrow = b * Sc + sb;
  const int hc = h * HDc;
  const int vrow = (b * NHc + h) * HDc;

  const bf16x16 Q0 = load_a_tile(qbf, Hc, qrow, hc, r, half);
  const bf16x16 Q1 = load_a_tile(qbf, Hc, qrow, hc + 32, r, half);

  // prologue: stage 0 K/V tiles
  if (wave == 0)
    tdm_load_2d_bf16(lds_addr_of(&Ksh[0][0]), kbf + (size_t)(b * Sc) * Hc + hc,
                     Hc, Mc, 64, 32);
  else if (wave == 1)
    tdm_load_2d_bf16(lds_addr_of(&Vsh[0][0]), vtb + (size_t)vrow * Sc,
                     Sc, Bc * NHc * HDc, 32, 64);
  if (wave < 2) tdm_wait();
  __syncthreads();

  f32x8 o[4];
#pragma unroll
  for (int j = 0; j < 4; ++j) o[j] = zero8();
  float mrun[8], lrun[8];
#pragma unroll
  for (int rr = 0; rr < 8; ++rr) { mrun[rr] = NEG_INF; lrun[rr] = 0.0f; }

  for (int kk = 0, stage = 0; kk < Sc; kk += 32, ++stage) {
    const int buf = stage & 1;
    // DMA next key block while we compute on this one
    if (kk + 32 < Sc) {
      if (wave == 0)
        tdm_load_2d_bf16(lds_addr_of(&Ksh[buf ^ 1][0]),
                         kbf + (size_t)(b * Sc + kk + 32) * Hc + hc, Hc, Mc, 64, 32);
      else if (wave == 1)
        tdm_load_2d_bf16(lds_addr_of(&Vsh[buf ^ 1][0]),
                         vtb + (size_t)vrow * Sc + kk + 32, Sc, Bc * NHc * HDc, 32, 64);
    }

    // ---- S = Q K^T for 16 queries x 32 keys (from LDS K tile) --------------
    f32x8 sc0 = zero8(), sc1 = zero8();
    {
      bf16x16 kb00 = load_b_tile(&Ksh[buf][0], 64, 0,  0,  r, half);
      bf16x16 kb01 = load_b_tile(&Ksh[buf][0], 64, 0,  32, r, half);
      bf16x16 kb10 = load_b_tile(&Ksh[buf][0], 64, 16, 0,  r, half);
      bf16x16 kb11 = load_b_tile(&Ksh[buf][0], 64, 16, 32, r, half);
      sc0 = wmma_bf16(Q0, kb00, sc0);
      sc0 = wmma_bf16(Q1, kb01, sc0);
      sc1 = wmma_bf16(Q0, kb10, sc1);
      sc1 = wmma_bf16(Q1, kb11, sc1);
    }

    // ---- mask + scale + online softmax (rows live per-VGPR across halves) --
    const bool keep0 = mask8[(size_t)b * Sc + kk + r] != 0;
    const bool keep1 = mask8[(size_t)b * Sc + kk + 16 + r] != 0;
    float corr[8];
#pragma unroll
    for (int rr = 0; rr < 8; ++rr) {
      float x0 = keep0 ? sc0[rr] * SCALE : NEG_INF;
      float x1 = keep1 ? sc1[rr] * SCALE : NEG_INF;
      float t = fmaxf(x0, x1);
      t = fmaxf(t, __shfl_xor(t, 1));
      t = fmaxf(t, __shfl_xor(t, 2));
      t = fmaxf(t, __shfl_xor(t, 4));
      t = fmaxf(t, __shfl_xor(t, 8));      // row max within 16-lane half
      const float mnew = fmaxf(mrun[rr], t);
      corr[rr] = __expf(mrun[rr] - mnew);
      mrun[rr] = mnew;
      const float p0 = __expf(x0 - mnew);
      const float p1 = __expf(x1 - mnew);
      sc0[rr] = p0; sc1[rr] = p1;
      float rs = p0 + p1;
      rs += __shfl_xor(rs, 1);
      rs += __shfl_xor(rs, 2);
      rs += __shfl_xor(rs, 4);
      rs += __shfl_xor(rs, 8);             // row sum within 16-lane half
      lrun[rr] = lrun[rr] * corr[rr] + rs;
    }
#pragma unroll
    for (int j = 0; j < 4; ++j)
#pragma unroll
      for (int rr = 0; rr < 8; ++rr) o[j][rr] *= corr[rr];

    // ---- relayout P from C-layout to A-layout through LDS -----------------
    u16* P = psh[wave];
#pragma unroll
    for (int rr = 0; rr < 8; ++rr) {
      const int row = rr + half * 8;
      P[row * 32 + r]      = f2bf(sc0[rr]);
      P[row * 32 + 16 + r] = f2bf(sc1[rr]);
    }
    __syncthreads();
    bf16x16 Pa;
    {
      const u16* base = P + r * 32 + half * 8;
      TileU t;
      t.q[0] = *(const int4*)(base);
      t.q[1] = *(const int4*)(base + 16);
      Pa = t.v;
    }

    // ---- O += P V  (V^T tile rows are contiguous in LDS) -------------------
#pragma unroll
    for (int j = 0; j < 4; ++j) {
      bf16x16 vt = load_b_tile(&Vsh[buf][0], 32, j * 16, 0, r, half);
      o[j] = wmma_bf16(Pa, vt, o[j]);
    }

    if (wave < 2) tdm_wait();   // next K/V tiles landed
    __syncthreads();            // readers done; next tiles visible to all
  }

  // ---- epilogue: normalize by row sum, store bf16 [B*S][H] -----------------
#pragma unroll
  for (int rr = 0; rr < 8; ++rr) {
    const int srow = sb + rr + half * 8;
    const size_t base = (size_t)(b * Sc + srow) * Hc + hc;
    const float inv = 1.0f / lrun[rr];
    obf[base + 0  + r] = f2bf(o[0][rr] * inv);
    obf[base + 16 + r] = f2bf(o[1][rr] * inv);
    obf[base + 32 + r] = f2bf(o[2][rr] * inv);
    obf[base + 48 + r] = f2bf(o[3][rr] * inv);
  }
}

// ---------------------------------------------------------------------------
// Kernel 4: output projection.  out = attn_o @ wo^T + bo  -> f32 [B*S][H]
// Same TDM-staged structure as the QKV GEMM.
// ---------------------------------------------------------------------------
__global__ __launch_bounds__(256) void out_gemm_kernel(
    const u16* __restrict__ abf, const u16* __restrict__ wbf,
    const float* __restrict__ bias, float* __restrict__ out) {
  __shared__ u16 Ash[2][128 * 32];
  __shared__ u16 Bsh[2][128 * 32];

  const int wave = threadIdx.x >> 5;
  const int lane = threadIdx.x & 31;
  const int r = lane & 15, half = lane >> 4;
  const int blkM = blockIdx.x * 128;
  const int blkN = blockIdx.y * 128;
  const int mb = (wave >> 1) * 32;
  const int nb = (wave & 1) * 64;

  if (wave == 0)
    tdm_load_2d_bf16(lds_addr_of(&Ash[0][0]), abf + (size_t)blkM * Hc, Hc, Mc, 32, 128);
  else if (wave == 1)
    tdm_load_2d_bf16(lds_addr_of(&Bsh[0][0]), wbf + (size_t)blkN * Hc, Hc, Hc, 32, 128);
  if (wave < 2) tdm_wait();
  __syncthreads();

  f32x8 acc[2][4];
#pragma unroll
  for (int i = 0; i < 2; ++i)
#pragma unroll
    for (int j = 0; j < 4; ++j) acc[i][j] = zero8();

  for (int kk = 0, stage = 0; kk < Hc; kk += 32, ++stage) {
    const int buf = stage & 1;
    if (kk + 32 < Hc) {
      if (wave == 0)
        tdm_load_2d_bf16(lds_addr_of(&Ash[buf ^ 1][0]),
                         abf + (size_t)blkM * Hc + kk + 32, Hc, Mc, 32, 128);
      else if (wave == 1)
        tdm_load_2d_bf16(lds_addr_of(&Bsh[buf ^ 1][0]),
                         wbf + (size_t)blkN * Hc + kk + 32, Hc, Hc, 32, 128);
    }

    bf16x16 a0 = load_a_tile(&Ash[buf][0], 32, mb, 0, r, half);
    bf16x16 a1 = load_a_tile(&Ash[buf][0], 32, mb + 16, 0, r, half);
#pragma unroll
    for (int j = 0; j < 4; ++j) {
      bf16x16 bt = load_b_tile(&Bsh[buf][0], 32, nb + j * 16, 0, r, half);
      acc[0][j] = wmma_bf16(a0, bt, acc[0][j]);
      acc[1][j] = wmma_bf16(a1, bt, acc[1][j]);
    }

    if (wave < 2) tdm_wait();
    __syncthreads();
  }

#pragma unroll
  for (int i = 0; i < 2; ++i)
#pragma unroll
    for (int j = 0; j < 4; ++j) {
      const int col = blkN + nb + j * 16 + r;
      const float bcol = bias[col];
#pragma unroll
      for (int rr = 0; rr < 8; ++rr) {
        const int row = blkM + mb + i * 16 + rr + half * 8;
        out[(size_t)row * Hc + col] = acc[i][j][rr] + bcol;
      }
    }
}

// ---------------------------------------------------------------------------
// Host-side launcher
// ---------------------------------------------------------------------------
extern "C" void kernel_launch(void* const* d_in, const int* in_sizes, int n_in,
                              void* d_out, int out_size, void* d_ws, size_t ws_size,
                              hipStream_t stream) {
  const float* x  = (const float*)d_in[0];
  // jnp bool arrays are 1 byte per element
  const unsigned char* mask = (const unsigned char*)d_in[1];
  const float* wq = (const float*)d_in[2];
  const float* bq = (const float*)d_in[3];
  const float* wk = (const float*)d_in[4];
  const float* bk = (const float*)d_in[5];
  const float* wv = (const float*)d_in[6];
  const float* bv = (const float*)d_in[7];
  const float* wo = (const float*)d_in[8];
  const float* bo = (const float*)d_in[9];
  float* out = (float*)d_out;

  const size_t MK = (size_t)Mc * Hc;   // 6.29M elems
  const size_t KK = (size_t)Hc * Hc;   // 0.59M elems

  u16* xbf  = (u16*)d_ws;              // [M][H]
  u16* wbf  = xbf + MK;                // [3][H][H]  (wq, wk, wv)
  u16* wobf = wbf + 3 * KK;            // [H][H]
  u16* qbf  = wobf + KK;               // [M][H]
  u16* kbf  = qbf + MK;                // [M][H]
  u16* vtb  = kbf + MK;                // [B][NH][HD][S]
  u16* obf  = vtb + MK;                // [M][H]  (attention output)

  auto cvt = [&](const float* s, u16* dst, size_t n) {
    const int n4 = (int)(n / 4);
    f32_to_bf16_kernel<<<dim3((n4 + 255) / 256), dim3(256), 0, stream>>>(s, dst, n4);
  };
  cvt(x,  xbf,          MK);
  cvt(wq, wbf,          KK);
  cvt(wk, wbf + KK,     KK);
  cvt(wv, wbf + 2 * KK, KK);
  cvt(wo, wobf,         KK);

  qkv_gemm_kernel<<<dim3(Mc / 128, Hc / 128, 3), dim3(256), 0, stream>>>(
      xbf, wbf, bq, bk, bv, qbf, kbf, vtb);

  attn_kernel<<<dim3(Sc / 16 / 8, NHc, Bc), dim3(256), 0, stream>>>(
      qbf, kbf, vtb, mask, obf);

  out_gemm_kernel<<<dim3(Mc / 128, Hc / 128, 1), dim3(256), 0, stream>>>(
      obf, wobf, bo, out);
}
